// GoogleAttention_59339268161900
// MI455X (gfx1250) — matrix operand order repaired and checked by reading, not compiled, not run on hardware
//
#include <hip/hip_runtime.h>
#include <hip/hip_bf16.h>
#include <stdint.h>

// ---------------------------------------------------------------------------
// Types / helpers for CDNA5 (gfx1250) wave32 WMMA
// ---------------------------------------------------------------------------
typedef __bf16 bf16;
typedef bf16  v16bf __attribute__((ext_vector_type(16)));
typedef float v8f   __attribute__((ext_vector_type(8)));

#define WMMA_BF16(a, b, c) \
  __builtin_amdgcn_wmma_f32_16x16x32_bf16(false, (a), false, (b), (short)0, (c), false, false)

// XOR-lane shuffle via ds_swizzle (group-of-32 mode: xor=offset[14:10], and=0x1f).
// Masks <= 8 stay inside each 16-lane half of the wave, matching the WMMA
// C-tile layout (rows 0-7 in lanes 0-15, rows 8-15 in lanes 16-31).
template <int PAT>
__device__ __forceinline__ float swz_xor(float v) {
  return __int_as_float(__builtin_amdgcn_ds_swizzle(__float_as_int(v), PAT));
}
__device__ __forceinline__ float redmax16(float v) {
  v = fmaxf(v, swz_xor<0x041f>(v));   // xor 1
  v = fmaxf(v, swz_xor<0x081f>(v));   // xor 2
  v = fmaxf(v, swz_xor<0x101f>(v));   // xor 4
  v = fmaxf(v, swz_xor<0x201f>(v));   // xor 8
  return v;
}
__device__ __forceinline__ float redsum16(float v) {
  v += swz_xor<0x041f>(v);
  v += swz_xor<0x081f>(v);
  v += swz_xor<0x101f>(v);
  v += swz_xor<0x201f>(v);
  return v;
}

// Problem constants
#define NB 2
#define NPOOL 4096           // 16*16*16 pooled voxels per batch
#define CIN 128
#define CF 16
#define CH 64
#define COUT 96              // 16(f)+16(g)+64(h)

// ---------------------------------------------------------------------------
// Kernel 1: fused 1x1x1 conv (x @ [Wf|Wg|Wh]) + 2x2x2 maxpool.
// One wave = 16 source voxels = 2 pooled voxels (rows 0-7 -> pv0, 8-15 -> pv1).
// C-tile layout makes the pool a per-lane max over the 8 accumulator regs.
// ---------------------------------------------------------------------------
__global__ __launch_bounds__(256) void ga_proj_pool(
    const float* __restrict__ x,
    const float* __restrict__ Wf, const float* __restrict__ biasf,
    const float* __restrict__ Wg, const float* __restrict__ biasg,
    const float* __restrict__ Wh, const float* __restrict__ biash,
    bf16* __restrict__ fbuf, bf16* __restrict__ gbuf, bf16* __restrict__ hbuf) {
  // Stage combined weights in LDS as bf16, column-major: WL[col*128 + ch]
  __shared__ bf16 WL[COUT * CIN];
  const int tid = threadIdx.x;
  for (int idx = tid; idx < COUT * CIN; idx += 256) {
    const int col = idx >> 7, ch = idx & 127;
    float v;
    if (col < 16)      v = Wf[ch * 16 + col];
    else if (col < 32) v = Wg[ch * 16 + (col - 16)];
    else               v = Wh[ch * 64 + (col - 32)];
    WL[idx] = (bf16)v;
  }
  __syncthreads();

  const int wave   = (blockIdx.x << 3) + (tid >> 5);  // 0..4095
  const int lane   = tid & 31;
  const int lanelo = lane & 15;
  const int half   = lane >> 4;

  // Pooled voxel this half-wave writes
  const int pv  = 2 * wave + half;
  const int b   = pv >> 12;
  const int pvb = pv & 4095;

  // Source voxel for this lane's A-matrix row (M = lane & 15, same for both halves)
  const int row   = lanelo;
  const int pvrow = 2 * wave + (row >> 3);
  const int brr = pvrow >> 12, pvbr = pvrow & 4095;
  const int phr = pvbr >> 8, pwr = (pvbr >> 4) & 15, pdr = pvbr & 15;
  const int r8 = row & 7;
  const int dh = (r8 >> 2) & 1, dw = (r8 >> 1) & 1, dd = r8 & 1;
  const float* xp = x + (size_t)brr * 4194304
                      + (size_t)(2 * phr + dh) * 131072
                      + (size_t)(2 * pwr + dw) * 4096
                      + (size_t)(2 * pdr + dd) * 128;

  v8f acc[6] = {};
  for (int kb = 0; kb < 4; ++kb) {
    const int cb = kb * 32;
    v16bf a;
#pragma unroll
    for (int t = 0; t < 8; ++t) {               // A 16x32 bf16 layout (wave32)
      a[t]     = (bf16)xp[cb + 16 * half + t];       // K = 16h..16h+7
      a[8 + t] = (bf16)xp[cb + 16 * half + 8 + t];   // K = 16h+8..16h+15
    }
#pragma unroll
    for (int t = 0; t < 6; ++t) {
      const int col  = t * 16 + lanelo;
      const bf16* wl = &WL[col * CIN + cb + 16 * half];
      v16bf bb;
#pragma unroll
      for (int j = 0; j < 16; ++j) bb[j] = wl[j];    // K = cb+16h+j
      acc[t] = WMMA_BF16(a, bb, acc[t]);
    }
  }

  // Per-lane max over the 8 C-rows = 2x2x2 maxpool; bias commutes with max.
#pragma unroll
  for (int t = 0; t < 6; ++t) {
    float mv = acc[t][0];
#pragma unroll
    for (int r = 1; r < 8; ++r) mv = fmaxf(mv, acc[t][r]);
    const int col = t * 16 + lanelo;
    const size_t base = (size_t)b * NPOOL + pvb;
    if (col < 16)
      fbuf[base * CF + col] = (bf16)(mv + biasf[col]);
    else if (col < 32)
      gbuf[base * CF + (col - 16)] = (bf16)(mv + biasg[col - 16]);
    else
      hbuf[base * CH + (col - 32)] = (bf16)(mv + biash[col - 32]);
  }
}

// ---------------------------------------------------------------------------
// Kernel 2: flash attention, one wave per 16-query tile (d_k=16 padded to 32,
// d_v=64). Online softmax; exp'd P transits a 2KB/wave LDS slab to get from
// C-tile layout into A-operand layout.
// ---------------------------------------------------------------------------
__global__ __launch_bounds__(64) void ga_flash_attn(
    const bf16* __restrict__ fbuf, const bf16* __restrict__ gbuf,
    const bf16* __restrict__ hbuf, float* __restrict__ o_ws) {
  __shared__ float p_lds[2][16 * 32];
  const int tid    = threadIdx.x;
  const int wl_id  = tid >> 5;
  const int lane   = tid & 31;
  const int lanelo = lane & 15;
  const int half   = lane >> 4;
  const int wave   = blockIdx.x * 2 + wl_id;     // 0..511
  const int b      = wave >> 8;
  const int qbase  = (wave & 255) << 4;

  // Q fragment: rows = queries, K = 16 real channels (K 16..31 zero-padded)
  const bf16* qp = gbuf + ((size_t)b * NPOOL + qbase + lanelo) * CF + 8 * half;
  v16bf qa;
#pragma unroll
  for (int t = 0; t < 8; ++t) { qa[t] = half == 0 ? qp[t] : (bf16)0.0f; qa[8 + t] = half == 0 ? qp[8 + t] : (bf16)0.0f; }

  v8f o0 = {}, o1 = {}, o2 = {}, o3 = {};
  float mrow[8], lrow[8];
#pragma unroll
  for (int r = 0; r < 8; ++r) { mrow[r] = -3.0e38f; lrow[r] = 0.0f; }

  float* pl = &p_lds[wl_id][0];
  const bf16* fb = fbuf + (size_t)b * NPOOL * CF;
  const bf16* hb = hbuf + (size_t)b * NPOOL * CH;

  for (int kb = 0; kb < NPOOL; kb += 32) {
    // S tiles: B operand is f^T (K=channel, N=key). Lanes with half==1 carry
    // the zero-padded channels 16..31.
    v16bf bf0 = {}, bf1 = {};
    if (half == 0) {
      const bf16* f0 = fb + (size_t)(kb + lanelo) * CF;
      const bf16* f1 = fb + (size_t)(kb + 16 + lanelo) * CF;
#pragma unroll
      for (int j = 0; j < 16; ++j) { bf0[j] = f0[j]; bf1[j] = f1[j]; }
    }
    const v8f z = {};
    v8f s0 = WMMA_BF16(qa, bf0, z);
    v8f s1 = WMMA_BF16(qa, bf1, z);

    // Online softmax update over this 32-key block
#pragma unroll
    for (int r = 0; r < 8; ++r) {
      const float a0 = s0[r], a1 = s1[r];
      const float rm = redmax16(fmaxf(a0, a1));
      const float nm = fmaxf(mrow[r], rm);
      const float sc = __expf(mrow[r] - nm);
      mrow[r] = nm;
      const float p0 = __expf(a0 - nm);
      const float p1 = __expf(a1 - nm);
      lrow[r] = lrow[r] * sc + redsum16(p0 + p1);
      o0[r] *= sc; o1[r] *= sc; o2[r] *= sc; o3[r] *= sc;
      const int prow = r + 8 * half;
      pl[prow * 32 + lanelo]      = p0;
      pl[prow * 32 + 16 + lanelo] = p1;
    }
    asm volatile("s_wait_dscnt 0" ::: "memory");  // wave-private LDS transpose

    // Repack P into A-operand layout (M = query row, K = 32 keys)
    const float* prow = pl + lanelo * 32;
    v16bf pa;
#pragma unroll
    for (int t = 0; t < 8; ++t) {
      pa[t]     = (bf16)prow[16 * half + t];
      pa[8 + t] = (bf16)prow[16 * half + 8 + t];
    }

    // O += P(16x32) @ H(32x64), four N-tiles of 16 channels
#pragma unroll
    for (int t = 0; t < 4; ++t) {
      v16bf hbf;
#pragma unroll
      for (int j = 0; j < 8; ++j) {
        const int key = kb + 2 * j + 16 * half;
        const int ch  = t * 16 + lanelo;
        hbf[2 * j]     = hb[(size_t)key * CH + ch];
        hbf[2 * j + 1] = hb[(size_t)(key + 1) * CH + ch];
      }
      if (t == 0)      o0 = WMMA_BF16(pa, hbf, o0);
      else if (t == 1) o1 = WMMA_BF16(pa, hbf, o1);
      else if (t == 2) o2 = WMMA_BF16(pa, hbf, o2);
      else             o3 = WMMA_BF16(pa, hbf, o3);
    }
  }

  float* op = o_ws + ((size_t)b * NPOOL + qbase) * CH;
#pragma unroll
  for (int r = 0; r < 8; ++r) {
    const int row   = r + 8 * half;
    const float inv = 1.0f / lrow[r];
    op[(size_t)row * CH +  0 + lanelo] = o0[r] * inv;
    op[(size_t)row * CH + 16 + lanelo] = o1[r] * inv;
    op[(size_t)row * CH + 32 + lanelo] = o2[r] * inv;
    op[(size_t)row * CH + 48 + lanelo] = o3[r] * inv;
  }
}

// ---------------------------------------------------------------------------
// Kernel 3: y = o_ws(8192x64) @ Wv(64x128) + bv, at pooled resolution.
// ---------------------------------------------------------------------------
__global__ __launch_bounds__(256) void ga_out_proj(
    const float* __restrict__ o_ws, const float* __restrict__ Wv,
    const float* __restrict__ bv, float* __restrict__ y) {
  __shared__ bf16 WL[128 * 64];  // column-major: WL[col*64 + ch]
  const int tid = threadIdx.x;
  for (int idx = tid; idx < 128 * 64; idx += 256) {
    const int col = idx >> 6, ch = idx & 63;
    WL[idx] = (bf16)Wv[ch * 128 + col];
  }
  __syncthreads();

  const int lane = tid & 31, lanelo = lane & 15, half = lane >> 4;
  const int wave = (blockIdx.x << 3) + (tid >> 5);  // 0..511
  const int rb   = wave << 4;

  const float* orow = o_ws + (size_t)(rb + lanelo) * CH;
  v8f acc[8] = {};
#pragma unroll
  for (int kb = 0; kb < 2; ++kb) {
    const int cb = kb * 32;
    v16bf a;
#pragma unroll
    for (int t = 0; t < 8; ++t) {
      a[t]     = (bf16)orow[cb + 16 * half + t];
      a[8 + t] = (bf16)orow[cb + 16 * half + 8 + t];
    }
#pragma unroll
    for (int t = 0; t < 8; ++t) {
      const int col  = t * 16 + lanelo;
      const bf16* wl = &WL[col * 64 + cb + 16 * half];
      v16bf bb;
#pragma unroll
      for (int j = 0; j < 16; ++j) bb[j] = wl[j];
      acc[t] = WMMA_BF16(a, bb, acc[t]);
    }
  }
#pragma unroll
  for (int t = 0; t < 8; ++t) {
    const int col = t * 16 + lanelo;
    const float bias = bv[col];
#pragma unroll
    for (int r = 0; r < 8; ++r) {
      const int row = rb + r + 8 * half;
      y[(size_t)row * 128 + col] = acc[t][r] + bias;
    }
  }
}

// ---------------------------------------------------------------------------
// Kernel 4: out = gamma * upsample2(y) + x   (float4, bandwidth bound)
// ---------------------------------------------------------------------------
__global__ __launch_bounds__(256) void ga_upsample_residual(
    const float* __restrict__ x, const float* __restrict__ y,
    const float* __restrict__ gamma, float* __restrict__ out) {
  const int idx = blockIdx.x * 256 + threadIdx.x;  // float4 index, 0..2097151
  const float g = gamma[0];
  const int c4 = idx & 31;
  const int vox = idx >> 5;
  const int d = vox & 31, w = (vox >> 5) & 31, h = (vox >> 10) & 31, b = vox >> 15;
  const int pooled = ((h >> 1) << 8) | ((w >> 1) << 4) | (d >> 1);
  const float4 xv = ((const float4*)x)[idx];
  const float4 yv = ((const float4*)y)[((size_t)b * NPOOL + pooled) * 32 + c4];
  float4 r;
  r.x = g * yv.x + xv.x; r.y = g * yv.y + xv.y;
  r.z = g * yv.z + xv.z; r.w = g * yv.w + xv.w;
  ((float4*)out)[idx] = r;
}

// ---------------------------------------------------------------------------
// Launch. Inputs (setup_inputs order): x, Wf, bf, Wg, bg, Wh, bh, Wv, bv, gamma
// Workspace (7.5 MB): f/g/h bf16, O f32, Y f32. Score matrix never materialized.
// ---------------------------------------------------------------------------
extern "C" void kernel_launch(void* const* d_in, const int* in_sizes, int n_in,
                              void* d_out, int out_size, void* d_ws, size_t ws_size,
                              hipStream_t stream) {
  (void)in_sizes; (void)n_in; (void)out_size; (void)ws_size;
  const float* x     = (const float*)d_in[0];
  const float* Wf    = (const float*)d_in[1];
  const float* biasf = (const float*)d_in[2];
  const float* Wg    = (const float*)d_in[3];
  const float* biasg = (const float*)d_in[4];
  const float* Wh    = (const float*)d_in[5];
  const float* biash = (const float*)d_in[6];
  const float* Wv    = (const float*)d_in[7];
  const float* bv    = (const float*)d_in[8];
  const float* gamma = (const float*)d_in[9];

  uint8_t* ws = (uint8_t*)d_ws;
  bf16*  fbuf = (bf16*)(ws);                              // 2*4096*16*2  = 256 KB
  bf16*  gbuf = (bf16*)(ws + (size_t)256 * 1024);         // 256 KB
  bf16*  hbuf = (bf16*)(ws + (size_t)512 * 1024);         // 2*4096*64*2  = 1 MB
  float* o_ws = (float*)(ws + (size_t)1536 * 1024);       // 2*4096*64*4  = 2 MB
  float* ybuf = (float*)(ws + (size_t)3584 * 1024);       // 2*4096*128*4 = 4 MB

  ga_proj_pool<<<512, 256, 0, stream>>>(x, Wf, biasf, Wg, biasg, Wh, biash,
                                        fbuf, gbuf, hbuf);
  ga_flash_attn<<<256, 64, 0, stream>>>(fbuf, gbuf, hbuf, o_ws);
  ga_out_proj<<<64, 256, 0, stream>>>(o_ws, Wv, bv, ybuf);
  ga_upsample_residual<<<8192, 256, 0, stream>>>(x, ybuf, gamma, (float*)d_out);
}
